// SAMAF_9517647528343
// MI455X (gfx1250) — compile-verified, matrix operand order checked by reading, not run on hardware
//
#include <hip/hip_runtime.h>

typedef __attribute__((ext_vector_type(16))) __bf16     v16bf;
typedef __attribute__((ext_vector_type(8)))  float      v8f;
typedef __attribute__((ext_vector_type(4)))  unsigned   v4u;

#define THREADS 128
#define NWAVES  4

static constexpr int F     = 13;    // input features
static constexpr int H     = 128;   // hidden
static constexpr int S     = 128;   // sequence length
static constexpr int NSEQ  = 2048;  // 64*32 flattened batch
static constexpr int KX    = 32;    // padded x width (13 -> 32)
static constexpr int KTOT  = 160;   // KX + H
static constexpr int STR   = 168;   // padded LDS row stride (bank-conflict-free)
static constexpr int FCSTR = 136;   // padded W_fc row stride
static constexpr int GATES = 512;   // 4*H

// LDS layout (bytes):
//   wcat : 512*STR ushort  (bf16 [W_ih|0|W_hh], row-major, stride STR)
//   wfc  : 16*FCSTR ushort (bf16 W_fc padded to 16 rows)
//   bsum : 512 float       (b_ih + b_hh)
//   bfcl : 16 float        (b_fc padded)
//   bufs : NWAVES * 2 * 16*STR ushort (double-buffered [x|h] per wave)
static constexpr size_t SHMEM_BYTES =
    (size_t)(GATES * STR + 16 * FCSTR) * 2 + (GATES + 16) * 4 +
    (size_t)NWAVES * 2 * 16 * STR * 2;  // = 221,504

static __device__ __forceinline__ unsigned short f2bf(float f) {
  unsigned u = __float_as_uint(f);
  u += 0x7FFFu + ((u >> 16) & 1u);        // round-to-nearest-even
  return (unsigned short)(u >> 16);
}
static __device__ __forceinline__ float bf2f(unsigned short s) {
  return __uint_as_float(((unsigned)s) << 16);
}
static __device__ __forceinline__ float sigm(float x) {
  return 1.0f / (1.0f + __expf(-x));
}
static __device__ __forceinline__ float tanh_(float x) {
  float e = __expf(-2.0f * x);
  return (1.0f - e) / (1.0f + e);
}

// Load one 16x32 bf16 fragment row-slice: lane holds row/col (lane&15),
// K chunks at k0..k0+7 and k0+16..k0+23 (k0 includes +8 for lanes 16..31).
// Two ds_load_b128 from a 16B-aligned, padded-stride LDS region.
static __device__ __forceinline__ v16bf ldfrag(const unsigned short* p) {
  v16bf r;
  ((v4u*)&r)[0] = *(const v4u*)(p);
  ((v4u*)&r)[1] = *(const v4u*)(p + 16);
  return r;
}

// One fused LSTM cell step for this wave's 16 rows.
// rb: read buffer [16 x STR] bf16, cols 0..31 = x (padded), 32..159 = h
// wb: write buffer; h_new goes to cols 32..159
static __device__ __forceinline__ void lstm_step(
    const unsigned short* __restrict__ rb, unsigned short* __restrict__ wb,
    const unsigned short* __restrict__ wcat, const float* __restrict__ bsum,
    v8f cst[8], int lane) {
  const int n   = lane & 15;
  const int hi8 = (lane & 16) ? 8 : 0;

  // A fragments ([x|h] rows of this wave) reused across all 32 gate tiles
  v16bf afr[5];
#pragma unroll
  for (int kc = 0; kc < 5; ++kc)
    afr[kc] = ldfrag(rb + n * STR + kc * 32 + hi8);

#pragma unroll
  for (int jt = 0; jt < 8; ++jt) {              // 8 tiles of 16 hidden cols
    v8f ai = {}, af = {}, ag = {}, ao = {};
    const unsigned short* wr = wcat + (jt * 16 + n) * STR;
#pragma unroll
    for (int kc = 0; kc < 5; ++kc) {            // K = 160 in chunks of 32
      const int k0 = kc * 32 + hi8;
      v16bf bi = ldfrag(wr + k0);               // gate i rows   0..127
      v16bf bff = ldfrag(wr + 128 * STR + k0);  // gate f rows 128..255
      v16bf bg = ldfrag(wr + 256 * STR + k0);   // gate g rows 256..383
      v16bf bo = ldfrag(wr + 384 * STR + k0);   // gate o rows 384..511
      ai = __builtin_amdgcn_wmma_f32_16x16x32_bf16(false, afr[kc], false, bi,
                                                   (short)0, ai, false, false);
      af = __builtin_amdgcn_wmma_f32_16x16x32_bf16(false, afr[kc], false, bff,
                                                   (short)0, af, false, false);
      ag = __builtin_amdgcn_wmma_f32_16x16x32_bf16(false, afr[kc], false, bg,
                                                   (short)0, ag, false, false);
      ao = __builtin_amdgcn_wmma_f32_16x16x32_bf16(false, afr[kc], false, bo,
                                                   (short)0, ao, false, false);
    }
    const int col = jt * 16 + n;
    const float b_i = bsum[col], b_f = bsum[128 + col];
    const float b_g = bsum[256 + col], b_o = bsum[384 + col];
#pragma unroll
    for (int r = 0; r < 8; ++r) {               // C layout: m = r + hi8, n = col
      float iv = sigm(ai[r] + b_i);
      float fv = sigm(af[r] + b_f);
      float gv = tanh_(ag[r] + b_g);
      float ov = sigm(ao[r] + b_o);
      float cn = fv * cst[jt][r] + iv * gv;
      cst[jt][r] = cn;
      wb[(r + hi8) * STR + KX + col] = f2bf(ov * tanh_(cn));
    }
  }
}

__global__ __launch_bounds__(THREADS, 1) void lstm_ae_wmma_kernel(
    const float* __restrict__ X, const float* __restrict__ Wih,
    const float* __restrict__ Whh, const float* __restrict__ bih,
    const float* __restrict__ bhh, const float* __restrict__ Wfc,
    const float* __restrict__ bfc, float* __restrict__ out) {
  extern __shared__ unsigned short smem[];
  unsigned short* wcat = smem;                          // 512*STR
  unsigned short* wfc  = wcat + GATES * STR;            // 16*FCSTR
  float* bsum = (float*)(wfc + 16 * FCSTR);             // 512
  float* bfcl = bsum + GATES;                           // 16
  unsigned short* bufs = (unsigned short*)(bfcl + 16);  // NWAVES*2*16*STR

  const int tid  = threadIdx.x;
  const int lane = tid & 31;
  const int wave = tid >> 5;
  const int n    = lane & 15;
  const int hi8  = (lane & 16) ? 8 : 0;

  // ---- stage fused weights [W_ih | 0 | W_hh] as bf16 ----
  for (int idx = tid; idx < GATES * KTOT; idx += THREADS) {
    int j = idx / KTOT, k = idx - j * KTOT;
    float v = (k < KX) ? ((k < F) ? Wih[j * F + k] : 0.0f) : Whh[j * H + (k - KX)];
    wcat[j * STR + k] = f2bf(v);
  }
  for (int idx = tid; idx < 16 * H; idx += THREADS) {
    int j = idx >> 7, k = idx & (H - 1);
    wfc[j * FCSTR + k] = f2bf((j < F) ? Wfc[j * H + k] : 0.0f);
  }
  for (int j = tid; j < GATES; j += THREADS) bsum[j] = bih[j] + bhh[j];
  for (int j = tid; j < 16; j += THREADS) bfcl[j] = (j < F) ? bfc[j] : 0.0f;
  for (int idx = tid; idx < NWAVES * 2 * 16 * STR; idx += THREADS) bufs[idx] = 0;
  __syncthreads();

  unsigned short* buf0 = bufs + wave * 2 * 16 * STR;
  unsigned short* buf1 = buf0 + 16 * STR;
  const int row0 = (blockIdx.x * NWAVES + wave) * 16;

  v8f cst[8];
#pragma unroll
  for (int jt = 0; jt < 8; ++jt) {
    v8f z = {};
    cst[jt] = z;
  }

  // ---- encoder: 128 sequential steps ----
  for (int t = 0; t < S; ++t) {
    unsigned short* rb = (t & 1) ? buf1 : buf0;
    unsigned short* wb = (t & 1) ? buf0 : buf1;
    // load x_t (fp32 -> bf16) into BOTH buffers (so x_{S-1} survives for decoder)
    for (int e = lane; e < 16 * F; e += 32) {
      int r = e / F, f = e - r * F;
      unsigned short v = f2bf(X[((row0 + r) * S + t) * F + f]);
      buf0[r * STR + f] = v;
      buf1[r * STR + f] = v;
    }
    lstm_step(rb, wb, wcat, bsum, cst, lane);
  }

  // ---- embeddings = h_n (buf0 h-region after step 127) ----
  for (int e = lane; e < 16 * H; e += 32) {
    int r = e >> 7, j = e & (H - 1);
    out[(row0 + r) * H + j] = bf2f(buf0[r * STR + KX + j]);
  }

  // ---- decoder: 128 sequential steps, out feeds next x ----
  float* dec = out + NSEQ * H;
  for (int t = 0; t < S; ++t) {
    unsigned short* rb = (t & 1) ? buf1 : buf0;
    unsigned short* wb = (t & 1) ? buf0 : buf1;
    lstm_step(rb, wb, wcat, bsum, cst, lane);
    // fc: out = h_new @ W_fc.T + b_fc   (one 16-col N-tile, K = 128)
    v8f acc = {};
#pragma unroll
    for (int kc = 0; kc < 4; ++kc) {
      const int k0 = kc * 32 + hi8;
      v16bf a = ldfrag(wb + n * STR + KX + k0);
      v16bf b = ldfrag(wfc + n * FCSTR + k0);
      acc = __builtin_amdgcn_wmma_f32_16x16x32_bf16(false, a, false, b,
                                                    (short)0, acc, false, false);
    }
    const float bb = bfcl[n];
    if (n < F) {
#pragma unroll
      for (int r = 0; r < 8; ++r) {
        int m = r + hi8;
        float v = acc[r] + bb;
        dec[((row0 + m) * S + t) * F + n] = v;   // decoder output
        wb[m * STR + n] = f2bf(v);               // next step's x (in next rb)
      }
    }
  }
}

extern "C" void kernel_launch(void* const* d_in, const int* in_sizes, int n_in,
                              void* d_out, int out_size, void* d_ws, size_t ws_size,
                              hipStream_t stream) {
  const float* X   = (const float*)d_in[0];
  const float* Wih = (const float*)d_in[1];
  const float* Whh = (const float*)d_in[2];
  const float* bih = (const float*)d_in[3];
  const float* bhh = (const float*)d_in[4];
  const float* Wfc = (const float*)d_in[5];
  const float* bfc = (const float*)d_in[6];
  float* out = (float*)d_out;

  (void)in_sizes; (void)n_in; (void)out_size; (void)d_ws; (void)ws_size;

  hipFuncSetAttribute((const void*)lstm_ae_wmma_kernel,
                      hipFuncAttributeMaxDynamicSharedMemorySize,
                      (int)SHMEM_BYTES);

  dim3 grid(NSEQ / (NWAVES * 16));  // 32 workgroups x 4 waves = 128 waves
  dim3 block(THREADS);
  lstm_ae_wmma_kernel<<<grid, block, SHMEM_BYTES, stream>>>(
      X, Wih, Whh, bih, bhh, Wfc, bfc, out);
}
